// HardNegativeContrastiveLoss_87900800680396
// MI455X (gfx1250) — matrix operand order repaired and checked by reading, not compiled, not run on hardware
//
#include <hip/hip_runtime.h>

typedef __attribute__((ext_vector_type(16))) __bf16 v16bf;
typedef __attribute__((ext_vector_type(8)))  __bf16 v8bf;
typedef __attribute__((ext_vector_type(8)))  float  v8f;

#define NROWS 8192
#define DIM   256
#define INV_TEMP 14.285714285714286f   // 1/0.07
#define LDA   264                      // padded LDS stride for A tile (bf16 elems)
#define SCOLS 512                      // columns per strip
#define SSTR  524                      // padded strip stride (floats)

__device__ __forceinline__ v16bf cat16(v8bf a, v8bf b) {
  return __builtin_shufflevector(a, b, 0,1,2,3,4,5,6,7,8,9,10,11,12,13,14,15);
}

// ---------------------------------------------------------------------------
// Split fp32 features into bf16 hi/lo pairs (hi + lo ~= x to ~fp32 accuracy)
// ---------------------------------------------------------------------------
__global__ __launch_bounds__(256) void split_kernel(
    const float* __restrict__ IF, const float* __restrict__ CF,
    __bf16* __restrict__ ifhi, __bf16* __restrict__ iflo,
    __bf16* __restrict__ cfhi, __bf16* __restrict__ cflo, int nd)
{
  int i = blockIdx.x * blockDim.x + threadIdx.x;
  if (i >= nd) return;
  float x = IF[i];
  __bf16 hx = (__bf16)x;
  ifhi[i] = hx;
  iflo[i] = (__bf16)(x - (float)hx);
  float y = CF[i];
  __bf16 hy = (__bf16)y;
  cfhi[i] = hy;
  cflo[i] = (__bf16)(y - (float)hy);
}

// ---------------------------------------------------------------------------
// Fused GEMM (bf16x3 split WMMA) + per-row top-32 + row loss.
// One workgroup (8 waves, 256 threads) owns 16 rows across all 8192 columns.
// ---------------------------------------------------------------------------
__global__ __launch_bounds__(256) void rowtopk_kernel(
    const __bf16* __restrict__ Ahi, const __bf16* __restrict__ Alo,
    const __bf16* __restrict__ Bhi, const __bf16* __restrict__ Blo,
    float* __restrict__ rowloss)
{
  __shared__ __bf16 lAhi[16 * LDA];
  __shared__ __bf16 lAlo[16 * LDA];
  __shared__ float  strip[16 * SSTR];
  __shared__ float  posv[16];

  const int tid  = threadIdx.x;
  const int lane = tid & 31;
  const int wave = tid >> 5;          // 0..7
  const int r0   = blockIdx.x * 16;   // first global row of this tile
  const int lrow = lane & 15;         // fragment row (A) / col (B) index
  const int half = lane >> 4;         // 0/1

  // ---- stage 16x256 A-tile (hi+lo) into padded LDS ----
  {
    int row = tid >> 4;               // 0..15
    int col = (tid & 15) * 16;        // 0..240
    const __bf16* gh = Ahi + (size_t)(r0 + row) * DIM + col;
    const __bf16* gl = Alo + (size_t)(r0 + row) * DIM + col;
    *(v8bf*)&lAhi[row * LDA + col]     = *(const v8bf*)gh;
    *(v8bf*)&lAhi[row * LDA + col + 8] = *(const v8bf*)(gh + 8);
    *(v8bf*)&lAlo[row * LDA + col]     = *(const v8bf*)gl;
    *(v8bf*)&lAlo[row * LDA + col + 8] = *(const v8bf*)(gl + 8);
  }
  __syncthreads();

  const float NEG_INF = -__builtin_inff();

  // per-row top-32, ascending (only meaningful in lanes tid<16)
  float topv[32];
#pragma unroll
  for (int i = 0; i < 32; ++i) topv[i] = NEG_INF;

  for (int s = 0; s < NROWS / SCOLS; ++s) {
    // ======== GEMM phase: each wave computes 4 16x16 tiles ========
#pragma unroll
    for (int t = 0; t < 4; ++t) {
      const int tileCol = s * (SCOLS / 16) + wave * 4 + t;   // 0..511
      const int c0      = tileCol * 16;
      const __bf16* bh = Bhi + (size_t)(c0 + lrow) * DIM + 8 * half;
      const __bf16* bl = Blo + (size_t)(c0 + lrow) * DIM + 8 * half;
      v8f acc = {};
#pragma unroll
      for (int k = 0; k < DIM; k += 32) {
        const int ka = k + 8 * half;
        // A fragments: chunks {ka..ka+7} and {ka+16..ka+23}
        v16bf ah = cat16(*(const v8bf*)&lAhi[lrow * LDA + ka],
                         *(const v8bf*)&lAhi[lrow * LDA + ka + 16]);
        v16bf al = cat16(*(const v8bf*)&lAlo[lrow * LDA + ka],
                         *(const v8bf*)&lAlo[lrow * LDA + ka + 16]);
        // B fragments: chunks {k+8*half..+7} and {k+16+8*half..+7}
        v16bf bhf = cat16(*(const v8bf*)(bh + k),
                          *(const v8bf*)(bh + k + 16));
        v16bf blf = cat16(*(const v8bf*)(bl + k),
                          *(const v8bf*)(bl + k + 16));
        acc = __builtin_amdgcn_wmma_f32_16x16x32_bf16(false, ah, false, bhf,
                                                      (short)0, acc, false, false);
        acc = __builtin_amdgcn_wmma_f32_16x16x32_bf16(false, ah, false, blf,
                                                      (short)0, acc, false, false);
        acc = __builtin_amdgcn_wmma_f32_16x16x32_bf16(false, al, false, bhf,
                                                      (short)0, acc, false, false);
      }
      // scale by 1/T, capture positive (diag), mask diag, spill to LDS strip
      const int colg   = c0 + lrow;
      const int stripc = (wave * 4 + t) * 16 + lrow;   // column within strip
#pragma unroll
      for (int v = 0; v < 8; ++v) {
        const int rowl = v + half * 8;
        float val = acc[v] * INV_TEMP;
        if (colg == r0 + rowl) { posv[rowl] = val; val = NEG_INF; }
        strip[rowl * SSTR + stripc] = val;
      }
    }
    __syncthreads();

    // ======== selection phase: lane-per-row running top-32 ========
    if (tid < 16) {
      const float* rowp = &strip[tid * SSTR];
      for (int c = 0; c < SCOLS; c += 4) {
        const float4 q = *(const float4*)&rowp[c];
        float vals[4] = {q.x, q.y, q.z, q.w};
#pragma unroll
        for (int u = 0; u < 4; ++u) {
          const float v = vals[u];
          if (v > topv[0]) {              // rare: ~180 hits per 8192 values
#pragma unroll
            for (int j = 0; j < 31; ++j) {
              const float nxt = topv[j + 1];
              topv[j] = (nxt <= v) ? nxt : ((topv[j] <= v) ? v : topv[j]);
            }
            topv[31] = (topv[31] <= v) ? v : topv[31];
          }
        }
      }
    }
    __syncthreads();
  }

  // ======== row loss: logsumexp([pos, top32]) - pos ========
  if (tid < 16) {
    const float p = posv[tid];
    float m = fmaxf(p, topv[31]);
    float ssum = __expf(p - m);
#pragma unroll
    for (int i = 0; i < 32; ++i) ssum += __expf(topv[i] - m);
    rowloss[r0 + tid] = m + __logf(ssum) - p;
  }
}

// ---------------------------------------------------------------------------
// Deterministic final reduction: (sum over 2*N row losses) / (2*N)
// ---------------------------------------------------------------------------
__global__ __launch_bounds__(256) void reduce_kernel(
    const float* __restrict__ rowloss, float* __restrict__ out, int n)
{
  __shared__ float sm[256];
  float s = 0.f;
  for (int i = threadIdx.x; i < n; i += 256) s += rowloss[i];
  sm[threadIdx.x] = s;
  __syncthreads();
  for (int w = 128; w > 0; w >>= 1) {
    if (threadIdx.x < w) sm[threadIdx.x] += sm[threadIdx.x + w];
    __syncthreads();
  }
  if (threadIdx.x == 0) out[0] = sm[0] / (float)n;
}

// ---------------------------------------------------------------------------
extern "C" void kernel_launch(void* const* d_in, const int* in_sizes, int n_in,
                              void* d_out, int out_size, void* d_ws, size_t ws_size,
                              hipStream_t stream) {
  const float* IF = (const float*)d_in[0];
  const float* CF = (const float*)d_in[1];
  float* out = (float*)d_out;

  const int N = NROWS, D = DIM, ND = N * D;

  // workspace layout: 4 bf16 arrays (hi/lo for both matrices) + row losses
  __bf16* ifhi = (__bf16*)d_ws;
  __bf16* iflo = ifhi + ND;
  __bf16* cfhi = iflo + ND;
  __bf16* cflo = cfhi + ND;
  float*  rowloss = (float*)(cflo + ND);

  split_kernel<<<(ND + 255) / 256, 256, 0, stream>>>(IF, CF, ifhi, iflo, cfhi, cflo, ND);

  // i2c: rows = image_features, cols = current_features
  rowtopk_kernel<<<N / 16, 256, 0, stream>>>(ifhi, iflo, cfhi, cflo, rowloss);
  // c2i: rows = current_features, cols = image_features (columns of logits)
  rowtopk_kernel<<<N / 16, 256, 0, stream>>>(cfhi, cflo, ifhi, iflo, rowloss + N);

  reduce_kernel<<<1, 256, 0, stream>>>(rowloss, out, 2 * N);
}
